// TokenClassificationWithCRF_57578331571010
// MI455X (gfx1250) — matrix operand order repaired and checked by reading, not compiled, not run on hardware
//
#include <hip/hip_runtime.h>

// ---------------------------------------------------------------------------
// Problem constants (from reference)
// ---------------------------------------------------------------------------
constexpr int Bb  = 16;
constexpr int Ss  = 512;
constexpr int Dd  = 768;
constexpr int Hh  = 12;
constexpr int Ll  = 4;
constexpr int Kk  = 9;
constexpr int Ff  = 3072;
constexpr int HDd = 64;          // head dim
constexpr int TD  = 3 * Dd;      // 2304
constexpr int BS  = Bb * Ss;     // 8192 tokens

typedef __bf16 bf16;
typedef __attribute__((ext_vector_type(16))) __bf16 v16bf;
typedef __attribute__((ext_vector_type(8)))  float  v8f;

union Frag16 { v16bf v; uint4 u[2]; };

// ---------------------------------------------------------------------------
// Elementwise helpers
// ---------------------------------------------------------------------------
__global__ void k_zero_f32(float* p, int n) {
    int i = blockIdx.x * blockDim.x + threadIdx.x;
    if (i < n) p[i] = 0.0f;
}

__global__ void k_f32_to_bf16(const float* __restrict__ src, bf16* __restrict__ dst, size_t n) {
    size_t i = (size_t)blockIdx.x * blockDim.x + threadIdx.x;
    size_t stride = (size_t)gridDim.x * blockDim.x;
    for (; i < n; i += stride) dst[i] = (bf16)src[i];
}

__global__ void k_gelu_f32_to_bf16(const float* __restrict__ src, bf16* __restrict__ dst, size_t n) {
    size_t i = (size_t)blockIdx.x * blockDim.x + threadIdx.x;
    size_t stride = (size_t)gridDim.x * blockDim.x;
    for (; i < n; i += stride) {
        float x = src[i];
        float t = 0.7978845608028654f * (x + 0.044715f * x * x * x);
        dst[i] = (bf16)(0.5f * x * (1.0f + tanhf(t)));
    }
}

// ---------------------------------------------------------------------------
// Embedding gather: out[row,:] = emb[ids[row],:] + pos[row%S,:]
// ---------------------------------------------------------------------------
__global__ void k_embed(const int* __restrict__ ids, const float* __restrict__ emb,
                        const float* __restrict__ pos, float* __restrict__ out) {
    int row = blockIdx.x;            // 0..BS-1
    int s   = row % Ss;
    int id  = ids[row];
    for (int c = threadIdx.x; c < Dd; c += blockDim.x)
        out[(size_t)row * Dd + c] = emb[(size_t)id * Dd + c] + pos[(size_t)s * Dd + c];
}

// ---------------------------------------------------------------------------
// LayerNorm over D=768 (optionally adding residual y): out = LN(x [+ y]) * g + b
// One block (256 threads) per token row. Safe for out == x.
// ---------------------------------------------------------------------------
__global__ void k_add_ln(const float* __restrict__ x, const float* __restrict__ y,
                         const float* __restrict__ g, const float* __restrict__ b,
                         float* __restrict__ out) {
    int row = blockIdx.x;
    int tid = threadIdx.x;
    __shared__ float r1[256], r2[256];
    float s = 0.f, s2 = 0.f;
    for (int c = tid; c < Dd; c += 256) {
        float v = x[(size_t)row * Dd + c] + (y ? y[(size_t)row * Dd + c] : 0.f);
        s += v; s2 += v * v;
    }
    r1[tid] = s; r2[tid] = s2;
    __syncthreads();
    for (int st = 128; st > 0; st >>= 1) {
        if (tid < st) { r1[tid] += r1[tid + st]; r2[tid] += r2[tid + st]; }
        __syncthreads();
    }
    float m    = r1[0] * (1.0f / Dd);
    float var  = r2[0] * (1.0f / Dd) - m * m;
    float rstd = rsqrtf(var + 1e-12f);
    for (int c = tid; c < Dd; c += 256) {
        float v = x[(size_t)row * Dd + c] + (y ? y[(size_t)row * Dd + c] : 0.f);
        out[(size_t)row * Dd + c] = (v - m) * rstd * g[c] + b[c];
    }
}

// ---------------------------------------------------------------------------
// bf16 WMMA GEMM: C[M,N] = A[M,Kd] * B[Kd,N] + bias[N]
// Block: 256 threads = 8 waves; block tile 128(M) x 64(N); K stage 64.
// Double-buffered LDS. A tile staged with CDNA5 async global->LDS DMA
// (global_load_async_to_lds_b128, ASYNCcnt-pipelined: next stage prefetches
// while current stage's 8 WMMAs run). B tile staged transposed (Bs[n][k]) via
// register transpose with paired-k 4B LDS stores, so both fragment gathers are
// contiguous 16B LDS vector loads matching the ISA WMMA VGPR layouts
// (A and B: 8-interleaved K halves).
// ---------------------------------------------------------------------------
template <bool OUT_BF16>
__global__ void k_gemm_bf16(const bf16* __restrict__ A, const bf16* __restrict__ Bw,
                            const float* __restrict__ bias,
                            float* __restrict__ Cf, bf16* __restrict__ Cb,
                            int M, int N, int Kd) {
    __shared__ alignas(16) bf16 As[2][128][64];   // 2 x 16 KB
    __shared__ alignas(16) bf16 Bs[2][64][64];    // 2 x  8 KB, transposed [n][k]

    const int tid  = threadIdx.x;
    const int wave = tid >> 5;
    const int lane = tid & 31;
    const int grp  = lane >> 4;     // 0/1: which lane-half
    const int l16  = lane & 15;
    const int m0   = blockIdx.y * 128;
    const int n0   = blockIdx.x * 64;

    // LDS byte offset of As: low 32 bits of the generic pointer are the LDS
    // address (ISA 10.2 aperture mapping).
    const unsigned asBase   = (unsigned)(size_t)(void*)&As[0][0][0];
    const unsigned asStride = 128u * 64u * 2u;   // bytes per buffer

    v8f acc[4] = {};

    // ---- A tile: 128x64 bf16 via async DMA; 4 x b128 per thread ----
    auto stageA_async = [&](int buf, int k0) {
        unsigned base = asBase + (unsigned)buf * asStride;
        #pragma unroll
        for (int it = 0; it < 4; ++it) {
            int idx = it * 256 + tid;            // 0..1023 uint4 slots
            int r   = idx >> 4;                  // row (16 x uint4 per row)
            int c   = (idx & 15) * 8;            // bf16 col
            unsigned lds = base + (unsigned)idx * 16u;
            unsigned long long ga =
                (unsigned long long)(const void*)&A[(size_t)(m0 + r) * Kd + k0 + c];
            asm volatile("global_load_async_to_lds_b128 %0, %1, off"
                         :: "v"(lds), "v"(ga) : "memory");
        }
    };
    // ---- B tile: 64(k)x64(n), transposed into Bs[n][k]; 2 rows of k paired ----
    auto stageB = [&](int buf, int k0) {
        int kk = (tid >> 3) * 2;                 // even k
        int nn = (tid & 7) * 8;
        uint4 va = *(const uint4*)&Bw[(size_t)(k0 + kk) * N + n0 + nn];
        uint4 vb = *(const uint4*)&Bw[(size_t)(k0 + kk + 1) * N + n0 + nn];
        const unsigned short* pa = (const unsigned short*)&va;
        const unsigned short* pb = (const unsigned short*)&vb;
        #pragma unroll
        for (int j = 0; j < 8; ++j) {
            unsigned pair = (unsigned)pa[j] | ((unsigned)pb[j] << 16);
            *(unsigned*)&Bs[buf][nn + j][kk] = pair;
        }
    };
    auto compute = [&](int buf) {
        #pragma unroll
        for (int sub = 0; sub < 2; ++sub) {      // two K=32 sub-steps
            Frag16 af;
            int row = wave * 16 + l16;
            af.u[0] = *(const uint4*)&As[buf][row][sub * 32 + 8 * grp];
            af.u[1] = *(const uint4*)&As[buf][row][sub * 32 + 16 + 8 * grp];
            #pragma unroll
            for (int t = 0; t < 4; ++t) {
                Frag16 bfg;
                int col = t * 16 + l16;
                bfg.u[0] = *(const uint4*)&Bs[buf][col][sub * 32 + 8 * grp];
                bfg.u[1] = *(const uint4*)&Bs[buf][col][sub * 32 + 8 * grp + 16];
                acc[t] = __builtin_amdgcn_wmma_f32_16x16x32_bf16(
                    false, af.v, false, bfg.v, (short)0, acc[t], false, false);
            }
        }
    };

    const int nsteps = Kd >> 6;                  // Kd/64 (768->12, 3072->48)
    stageA_async(0, 0);
    stageB(0, 0);
    for (int s = 0; s < nsteps; ++s) {
        int buf = s & 1;
        if (s + 1 < nsteps) {
            stageA_async(buf ^ 1, (s + 1) * 64); // prefetch next stage (async)
            stageB(buf ^ 1, (s + 1) * 64);
            // 8 async ops outstanding; <=4 releases current buffer (in-order)
            asm volatile("s_wait_asynccnt 0x4" ::: "memory");
        } else {
            asm volatile("s_wait_asynccnt 0x0" ::: "memory");
        }
        __syncthreads();
        compute(buf);
        __syncthreads();                         // protect buf before re-stage
    }

    // --- epilogue: C/D layout: VGPR r -> M = r + 8*grp, N = l16 ---
    #pragma unroll
    for (int t = 0; t < 4; ++t) {
        int col  = n0 + t * 16 + l16;
        float bv = bias ? bias[col] : 0.f;
        #pragma unroll
        for (int r = 0; r < 8; ++r) {
            int row = m0 + wave * 16 + r + 8 * grp;
            float v = acc[t][r] + bv;
            if constexpr (OUT_BF16) Cb[(size_t)row * N + col] = (bf16)v;
            else                    Cf[(size_t)row * N + col] = v;
        }
    }
}

// ---------------------------------------------------------------------------
// Fused attention for one (b, h, 16-query tile). Block = 128 threads = 4 waves.
// Phase 1: S_tile = (Q K^T)/8 + maskbias  -> LDS fp32 (16 x 512)
// Phase 2: softmax rows -> bf16 P in LDS  (16 x 512)
// Phase 3: ctx = P V ; each wave owns one 16-wide slice of d (64 total).
// qkv layout: [b, s, 3D] bf16; q at +0, k at +D, v at +2D within the 3D row.
// ---------------------------------------------------------------------------
__global__ void k_attention(const bf16* __restrict__ qkv, const int* __restrict__ amask,
                            float* __restrict__ ctx) {
    const int qt = blockIdx.x, h = blockIdx.y, b = blockIdx.z;
    const int tid = threadIdx.x, wave = tid >> 5, lane = tid & 31;
    const int grp = lane >> 4, l16 = lane & 15;

    __shared__ alignas(16) float sc[16][Ss];   // 32 KB scores
    __shared__ alignas(16) bf16  pb[16][Ss];   // 16 KB probs (bf16)
    __shared__ alignas(16) bf16  vt[64][32];   //  4 KB V chunk, transposed [d][key]

    // Q fragments (A role): row = query l16, two K-chunks of 32 along d
    Frag16 qf[2];
    {
        int q = qt * 16 + l16;
        const bf16* qp = qkv + (size_t)(b * Ss + q) * TD + h * HDd;
        #pragma unroll
        for (int c = 0; c < 2; ++c) {
            qf[c].u[0] = *(const uint4*)(qp + c * 32 + 8 * grp);
            qf[c].u[1] = *(const uint4*)(qp + c * 32 + 8 * grp + 16);
        }
    }

    // ---- Phase 1: scores; wave handles 8 of 32 key tiles ----
    for (int j = 0; j < 8; ++j) {
        int kt   = wave * 8 + j;
        int kcol = kt * 16 + l16;
        const bf16* kp = qkv + (size_t)(b * Ss + kcol) * TD + Dd + h * HDd;
        v8f acc = {};
        #pragma unroll
        for (int c = 0; c < 2; ++c) {
            Frag16 kf;          // B role: 8-interleaved d halves per lane-half
            kf.u[0] = *(const uint4*)(kp + c * 32 + 8 * grp);
            kf.u[1] = *(const uint4*)(kp + c * 32 + 8 * grp + 16);
            acc = __builtin_amdgcn_wmma_f32_16x16x32_bf16(
                false, qf[c].v, false, kf.v, (short)0, acc, false, false);
        }
        float mb = (1.0f - (float)amask[b * Ss + kcol]) * -1.0e9f;
        #pragma unroll
        for (int r = 0; r < 8; ++r)
            sc[r + 8 * grp][kcol] = acc[r] * 0.125f + mb;   // 1/sqrt(64)
    }
    __syncthreads();

    // ---- Phase 2: softmax; 8 threads per row, reductions via shfl ----
    {
        int row = tid >> 3, sub = tid & 7;
        float mx = -3.0e38f;
        for (int c = sub; c < Ss; c += 8) mx = fmaxf(mx, sc[row][c]);
        for (int o = 1; o < 8; o <<= 1) mx = fmaxf(mx, __shfl_xor(mx, o, 32));
        float sm = 0.f;
        for (int c = sub; c < Ss; c += 8) sm += expf(sc[row][c] - mx);
        for (int o = 1; o < 8; o <<= 1) sm += __shfl_xor(sm, o, 32);
        float inv = 1.0f / sm;
        for (int c = sub; c < Ss; c += 8) pb[row][c] = (bf16)(expf(sc[row][c] - mx) * inv);
    }

    // ---- Phase 3: ctx = P V ; wave w owns d slice [w*16, w*16+16) ----
    v8f cacc = {};
    for (int kc = 0; kc < 16; ++kc) {
        __syncthreads();   // also covers pb-completion before first stage
        // stage V chunk transposed: vt[d][key], 32 keys x 64 d
        for (int it = tid; it < 256; it += 128) {
            int key = it >> 3, db = (it & 7) * 8;
            const bf16* vp = qkv + (size_t)(b * Ss + kc * 32 + key) * TD + 2 * Dd + h * HDd + db;
            uint4 tv = *(const uint4*)vp;
            const bf16* tt = (const bf16*)&tv;
            #pragma unroll
            for (int j = 0; j < 8; ++j) vt[db + j][key] = tt[j];
        }
        __syncthreads();
        Frag16 pf, vf;
        pf.u[0] = *(const uint4*)&pb[l16][kc * 32 + 8 * grp];       // A role
        pf.u[1] = *(const uint4*)&pb[l16][kc * 32 + 8 * grp + 16];
        int dcol = wave * 16 + l16;
        vf.u[0] = *(const uint4*)&vt[dcol][8 * grp];                // B role
        vf.u[1] = *(const uint4*)&vt[dcol][8 * grp + 16];
        cacc = __builtin_amdgcn_wmma_f32_16x16x32_bf16(
            false, pf.v, false, vf.v, (short)0, cacc, false, false);
    }
    {
        int dcol = wave * 16 + l16;
        #pragma unroll
        for (int r = 0; r < 8; ++r) {
            int q = qt * 16 + r + 8 * grp;
            ctx[(size_t)(b * Ss + q) * Dd + h * HDd + dcol] = cacc[r];
        }
    }
}

// ---------------------------------------------------------------------------
// Classifier: logits[token,k] = x[token,:] . Wcls[:,k] + bcls[k]   (D x 9)
// ---------------------------------------------------------------------------
__global__ void k_classifier(const float* __restrict__ x, const float* __restrict__ W,
                             const float* __restrict__ bias, float* __restrict__ logits) {
    int idx = blockIdx.x * blockDim.x + threadIdx.x;
    if (idx >= BS * Kk) return;
    int token = idx / Kk, k = idx % Kk;
    float acc = bias[k];
    for (int d = 0; d < Dd; ++d)
        acc += x[(size_t)token * Dd + d] * W[(size_t)d * Kk + k];
    logits[idx] = acc;
}

// ---------------------------------------------------------------------------
// CRF: per batch (one wave): path score + forward logsumexp; out += logZ-score
// ---------------------------------------------------------------------------
__global__ void k_crf(const float* __restrict__ logits, const int* __restrict__ amask,
                      const int* __restrict__ labels, const float* __restrict__ start_t,
                      const float* __restrict__ end_t, const float* __restrict__ trans,
                      float* __restrict__ out) {
    const int b = blockIdx.x;
    const int lane = threadIdx.x;
    const float* lg  = logits + (size_t)b * Ss * Kk;
    const int*   lab = labels + b * Ss;
    const int*   mk  = amask  + b * Ss;

    // ---- gold path score ----
    float partial = 0.f, msum = 0.f;
    for (int s = lane; s < Ss; s += 32) {
        float em = lg[s * Kk + lab[s]];
        if (s == 0) partial += start_t[lab[0]] + em;
        else        partial += (float)mk[s] * (trans[lab[s - 1] * Kk + lab[s]] + em);
        msum += (float)mk[s];
    }
    for (int o = 16; o > 0; o >>= 1) {
        partial += __shfl_down(partial, o, 32);
        msum    += __shfl_down(msum, o, 32);
    }
    partial = __shfl(partial, 0, 32);
    msum    = __shfl(msum, 0, 32);
    int last = (int)msum - 1;
    float score = partial + end_t[lab[last]];

    // ---- forward algorithm (lane j holds alpha_j; K=9 active) ----
    int j  = lane;
    int jj = (j < Kk) ? j : 0;
    float alpha = start_t[jj] + lg[jj];
    for (int s = 1; s < Ss; ++s) {
        float e  = lg[s * Kk + jj];
        float mx = -3.0e38f;
        for (int i = 0; i < Kk; ++i)
            mx = fmaxf(mx, __shfl(alpha, i, 32) + trans[i * Kk + jj]);
        float sm = 0.f;
        for (int i = 0; i < Kk; ++i)
            sm += expf(__shfl(alpha, i, 32) + trans[i * Kk + jj] - mx);
        float nxt = mx + logf(sm) + e;
        if (mk[s] > 0) alpha = nxt;
    }
    float val = (j < Kk) ? (alpha + end_t[jj]) : -3.0e38f;
    float mx = val;
    for (int o = 16; o > 0; o >>= 1) mx = fmaxf(mx, __shfl_xor(mx, o, 32));
    float sm = (j < Kk) ? expf(val - mx) : 0.f;
    for (int o = 16; o > 0; o >>= 1) sm += __shfl_xor(sm, o, 32);
    float logZ = mx + logf(sm);
    if (lane == 0) atomicAdd(out, logZ - score);
}

// ---------------------------------------------------------------------------
// Host orchestration
// ---------------------------------------------------------------------------
extern "C" void kernel_launch(void* const* d_in, const int* in_sizes, int n_in,
                              void* d_out, int out_size, void* d_ws, size_t ws_size,
                              hipStream_t stream) {
    const int*   ids     = (const int*)d_in[0];
    const int*   amask   = (const int*)d_in[1];
    const int*   labels  = (const int*)d_in[2];
    const float* emb     = (const float*)d_in[3];
    const float* pos     = (const float*)d_in[4];
    const float* lng_e   = (const float*)d_in[5];
    const float* lnb_e   = (const float*)d_in[6];
    const float* Wqkv    = (const float*)d_in[7];
    const float* bqkv    = (const float*)d_in[8];
    const float* Wo      = (const float*)d_in[9];
    const float* bo      = (const float*)d_in[10];
    const float* ln1g    = (const float*)d_in[11];
    const float* ln1b    = (const float*)d_in[12];
    const float* W1      = (const float*)d_in[13];
    const float* b1      = (const float*)d_in[14];
    const float* W2      = (const float*)d_in[15];
    const float* b2      = (const float*)d_in[16];
    const float* ln2g    = (const float*)d_in[17];
    const float* ln2b    = (const float*)d_in[18];
    const float* Wcls    = (const float*)d_in[19];
    const float* bcls    = (const float*)d_in[20];
    const float* start_t = (const float*)d_in[21];
    const float* end_t   = (const float*)d_in[22];
    const float* trans   = (const float*)d_in[23];
    float* out = (float*)d_out;

    // workspace carve-out (256B aligned)
    char* p = (char*)d_ws;
    auto alloc = [&](size_t bytes) -> void* {
        void* r = (void*)p;
        p += (bytes + 255) & ~(size_t)255;
        return r;
    };
    float* xf     = (float*)alloc((size_t)BS * Dd * 4);
    bf16*  xb     = (bf16*) alloc((size_t)BS * Dd * 2);
    bf16*  qkvb   = (bf16*) alloc((size_t)BS * TD * 2);
    float* ctxf   = (float*)alloc((size_t)BS * Dd * 4);
    bf16*  ctxb   = (bf16*) alloc((size_t)BS * Dd * 2);
    float* attnf  = (float*)alloc((size_t)BS * Dd * 4);
    float* hf     = (float*)alloc((size_t)BS * Ff * 4);
    bf16*  hb     = (bf16*) alloc((size_t)BS * Ff * 2);
    float* logits = (float*)alloc((size_t)BS * Kk * 4);
    bf16*  wqkvb  = (bf16*) alloc((size_t)Ll * Dd * TD * 2);
    bf16*  wob    = (bf16*) alloc((size_t)Ll * Dd * Dd * 2);
    bf16*  w1b    = (bf16*) alloc((size_t)Ll * Dd * Ff * 2);
    bf16*  w2b    = (bf16*) alloc((size_t)Ll * Ff * Dd * 2);
    (void)ws_size; (void)in_sizes; (void)n_in;

    auto cvt = [&](const float* s, bf16* d, size_t n) {
        int blocks = (int)((n + 255) / 256);
        if (blocks > 4096) blocks = 4096;
        k_f32_to_bf16<<<blocks, 256, 0, stream>>>(s, d, n);
    };

    // weights -> bf16 (deterministic per call)
    cvt(Wqkv, wqkvb, (size_t)Ll * Dd * TD);
    cvt(Wo,   wob,   (size_t)Ll * Dd * Dd);
    cvt(W1,   w1b,   (size_t)Ll * Dd * Ff);
    cvt(W2,   w2b,   (size_t)Ll * Ff * Dd);

    // embeddings + embedding LN
    k_embed<<<BS, 256, 0, stream>>>(ids, emb, pos, ctxf);
    k_add_ln<<<BS, 256, 0, stream>>>(ctxf, nullptr, lng_e, lnb_e, xf);

    for (int l = 0; l < Ll; ++l) {
        // QKV projection (bf16 out for attention)
        cvt(xf, xb, (size_t)BS * Dd);
        k_gemm_bf16<true><<<dim3(TD / 64, BS / 128), 256, 0, stream>>>(
            xb, wqkvb + (size_t)l * Dd * TD, bqkv + (size_t)l * TD,
            nullptr, qkvb, BS, TD, Dd);

        // attention -> ctxf
        k_attention<<<dim3(Ss / 16, Hh, Bb), 128, 0, stream>>>(qkvb, amask, ctxf);

        // output projection
        cvt(ctxf, ctxb, (size_t)BS * Dd);
        k_gemm_bf16<false><<<dim3(Dd / 64, BS / 128), 256, 0, stream>>>(
            ctxb, wob + (size_t)l * Dd * Dd, bo + (size_t)l * Dd,
            attnf, nullptr, BS, Dd, Dd);

        // residual + LN1 (in place on xf)
        k_add_ln<<<BS, 256, 0, stream>>>(xf, attnf, ln1g + (size_t)l * Dd,
                                         ln1b + (size_t)l * Dd, xf);

        // FFN
        cvt(xf, xb, (size_t)BS * Dd);
        k_gemm_bf16<false><<<dim3(Ff / 64, BS / 128), 256, 0, stream>>>(
            xb, w1b + (size_t)l * Dd * Ff, b1 + (size_t)l * Ff,
            hf, nullptr, BS, Ff, Dd);
        {
            size_t n = (size_t)BS * Ff;
            int blocks = (int)((n + 255) / 256); if (blocks > 8192) blocks = 8192;
            k_gelu_f32_to_bf16<<<blocks, 256, 0, stream>>>(hf, hb, n);
        }
        k_gemm_bf16<false><<<dim3(Dd / 64, BS / 128), 256, 0, stream>>>(
            hb, w2b + (size_t)l * Ff * Dd, b2 + (size_t)l * Dd,
            attnf, nullptr, BS, Dd, Ff);

        // residual + LN2 (in place on xf)
        k_add_ln<<<BS, 256, 0, stream>>>(xf, attnf, ln2g + (size_t)l * Dd,
                                         ln2b + (size_t)l * Dd, xf);
    }

    // classifier logits (D x 9: bandwidth-trivial, plain kernel)
    k_classifier<<<(BS * Kk + 255) / 256, 256, 0, stream>>>(xf, Wcls, bcls, logits);

    // CRF loss -> scalar
    k_zero_f32<<<(out_size + 255) / 256, 256, 0, stream>>>(out, out_size);
    k_crf<<<Bb, 32, 0, stream>>>(logits, amask, labels, start_t, end_t, trans, out);
}
